// MultiHeadSelfAttention_32066225832261
// MI455X (gfx1250) — matrix operand order repaired and checked by reading, not compiled, not run on hardware
//
#include <hip/hip_runtime.h>
#include <hip/hip_bf16.h>

// ---------------------------------------------------------------- constants
constexpr int BB = 2;      // batch
constexpr int SS = 2048;   // sequence
constexpr int DD = 1024;   // d_model
constexpr int HH = 16;     // heads
constexpr int DK = 64;     // d_k
constexpr float INV_SCALE = 0.125f;  // 1/sqrt(64)

// ---------------------------------------------------------------- types
typedef __bf16 bf16;
typedef __attribute__((ext_vector_type(16))) __bf16 v16bf;
typedef __attribute__((ext_vector_type(8)))  __bf16 v8bf;
typedef __attribute__((ext_vector_type(4)))  __bf16 v4bf;
typedef __attribute__((ext_vector_type(8)))  float  v8f;

__device__ __forceinline__ v16bf cat8(v8bf lo, v8bf hi) {
    union { v16bf v; v8bf h[2]; } u;
    u.h[0] = lo; u.h[1] = hi;
    return u.v;
}

__device__ __forceinline__ v8f wmma_bf16(v16bf a, v16bf b, v8f c) {
    // D = A(16x32 bf16) * B(32x16 bf16) + C(16x16 f32)
    return __builtin_amdgcn_wmma_f32_16x16x32_bf16(
        false, a, false, b, (short)0, c, false, false);
}

__device__ __forceinline__ v16bf f16x4_to_a(float4 f0, float4 f1, float4 f2, float4 f3) {
    v16bf a;
    a[0]  = (bf16)f0.x; a[1]  = (bf16)f0.y; a[2]  = (bf16)f0.z; a[3]  = (bf16)f0.w;
    a[4]  = (bf16)f1.x; a[5]  = (bf16)f1.y; a[6]  = (bf16)f1.z; a[7]  = (bf16)f1.w;
    a[8]  = (bf16)f2.x; a[9]  = (bf16)f2.y; a[10] = (bf16)f2.z; a[11] = (bf16)f2.w;
    a[12] = (bf16)f3.x; a[13] = (bf16)f3.y; a[14] = (bf16)f3.z; a[15] = (bf16)f3.w;
    return a;
}

// async copy 16B global -> LDS (ASYNCcnt-tracked, no VGPR round trip)
__device__ __forceinline__ void async_b128_to_lds(unsigned lds_byte_off,
                                                  unsigned long long gaddr) {
    asm volatile("global_load_async_to_lds_b128 %0, %1, off"
                 :: "v"(lds_byte_off), "v"(gaddr) : "memory");
}
__device__ __forceinline__ void wait_asynccnt0() {
    asm volatile("s_wait_asynccnt 0x0" ::: "memory");
}

// ---------------------------------------------------------------- fp32 -> bf16
__global__ void cvt_f32_bf16_kernel(const float* __restrict__ x,
                                    bf16* __restrict__ y, int n4) {
    int i = blockIdx.x * blockDim.x + threadIdx.x;
    if (i >= n4) return;
    float4 f = ((const float4*)x)[i];
    v4bf o;
    o[0] = (bf16)f.x; o[1] = (bf16)f.y; o[2] = (bf16)f.z; o[3] = (bf16)f.w;
    ((v4bf*)y)[i] = o;
}

// ---------------------------------------------------------------- GEMM  C = A * B^T
// A: [M,K] bf16 row-major.  B: [N,K] bf16 row-major.
// Block = 8 waves sharing one 16-row A strip: staged once into LDS with
// global_load_async_to_lds_b128, then each wave computes a 16x64 tile reading
// A fragments from LDS (rows padded +8 elems -> b128 reads hit 64 banks).
// Requires K == 1024 here (both call sites).
template <typename OutT>
__global__ void gemm_abT_kernel(const bf16* __restrict__ A,
                                const bf16* __restrict__ Bm,
                                OutT* __restrict__ C,
                                int M, int N, int K) {
    constexpr int KMAX = 1024;
    constexpr int LDA = KMAX + 8;          // padded row, 2064 B
    __shared__ bf16 sA[16][LDA];           // 33 KB

    int tid = threadIdx.x;
    int w = tid >> 5;
    int lane = tid & 31;
    int ntiles = N >> 6;                   // 64-col tiles
    int blocksPerM = ntiles >> 3;          // 8 waves per block
    int tm = blockIdx.x / blocksPerM;
    int tn = (blockIdx.x % blocksPerM) * 8 + w;
    if (tm * 16 >= M) return;
    int l15 = lane & 15, hi = lane >> 4;

    // ---- stage A strip (16 x K bf16) into LDS via async copies ----------
    const bf16* abase = A + (size_t)(tm * 16) * K;
    int chunksPerRow = K >> 3;             // 16B chunks per row (128)
    for (int chunk = tid; chunk < 16 * chunksPerRow; chunk += 256) {
        int row = chunk / chunksPerRow;
        int c8  = (chunk % chunksPerRow) * 8;   // element offset
        unsigned loff = (unsigned)(uintptr_t)(void*)&sA[row][c8];
        unsigned long long g =
            (unsigned long long)(uintptr_t)(abase + (size_t)row * K + c8);
        async_b128_to_lds(loff, g);
    }
    wait_asynccnt0();
    __syncthreads();

    const bf16* brow0 = Bm + (size_t)(tn * 64 + 0 * 16 + l15) * K + hi * 16;
    const bf16* brow1 = Bm + (size_t)(tn * 64 + 1 * 16 + l15) * K + hi * 16;
    const bf16* brow2 = Bm + (size_t)(tn * 64 + 2 * 16 + l15) * K + hi * 16;
    const bf16* brow3 = Bm + (size_t)(tn * 64 + 3 * 16 + l15) * K + hi * 16;
    v8f acc[4] = {v8f{}, v8f{}, v8f{}, v8f{}};

    for (int k0 = 0; k0 < K; k0 += 32) {
        // A fragment from LDS (ds_load_b128 x2)
        v16bf a = cat8(*(const v8bf*)(&sA[l15][k0 + hi * 16]),
                       *(const v8bf*)(&sA[l15][k0 + hi * 16 + 8]));
        v16bf b0 = cat8(*(const v8bf*)(brow0 + k0), *(const v8bf*)(brow0 + k0 + 8));
        v16bf b1 = cat8(*(const v8bf*)(brow1 + k0), *(const v8bf*)(brow1 + k0 + 8));
        v16bf b2 = cat8(*(const v8bf*)(brow2 + k0), *(const v8bf*)(brow2 + k0 + 8));
        v16bf b3 = cat8(*(const v8bf*)(brow3 + k0), *(const v8bf*)(brow3 + k0 + 8));
        if (k0 + 32 < K)
            __builtin_prefetch(brow0 + k0 + 32, 0, 0);
        acc[0] = wmma_bf16(a, b0, acc[0]);
        acc[1] = wmma_bf16(a, b1, acc[1]);
        acc[2] = wmma_bf16(a, b2, acc[2]);
        acc[3] = wmma_bf16(a, b3, acc[3]);
    }
#pragma unroll
    for (int t = 0; t < 4; ++t)
#pragma unroll
        for (int r = 0; r < 8; ++r) {
            int row = tm * 16 + r + hi * 8;
            int col = tn * 64 + t * 16 + l15;
            C[(size_t)row * N + col] = (OutT)acc[t][r];
        }
}

// ---------------------------------------------------------------- V^T per head:
// Vt[bh][d][s] = Vp[(b*S+s)*D + h*DK + d]
__global__ void transpose_v_kernel(const bf16* __restrict__ Vp,
                                   bf16* __restrict__ Vt) {
    size_t i = (size_t)blockIdx.x * blockDim.x + threadIdx.x;  // over B*H*DK*S
    int s = (int)(i % SS);
    size_t j = i / SS;
    int d = (int)(j % DK);
    int bh = (int)(j / DK);
    int b = bh >> 4, h = bh & 15;
    Vt[i] = Vp[(size_t)(b * SS + s) * DD + h * DK + d];
}

// ---------------------------------------------------------------- fused attention
// One 256-thread block (8 waves) owns (b, h, 16 query rows).
// LDS: sS[16][2048] fp32 score strip (128 KB) + sC[16][64] context accumulator.
__global__ void fused_attn_kernel(const bf16* __restrict__ Qp,
                                  const bf16* __restrict__ Kp,
                                  const bf16* __restrict__ Vt,
                                  float* __restrict__ attn,
                                  bf16* __restrict__ ctx) {
    extern __shared__ float smem[];
    float* sS = smem;                 // [16][SS]
    float* sC = smem + 16 * SS;       // [16][64]

    int tid = threadIdx.x;
    int w = tid >> 5;                 // wave 0..7
    int lane = tid & 31;
    int l15 = lane & 15, hi = lane >> 4;

    int qt = blockIdx.x & ((SS / 16) - 1);
    int bh = blockIdx.x >> 7;         // SS/16 == 128
    int b = bh >> 4, h = bh & 15;
    int q0 = qt * 16;

    // zero context accumulator
#pragma unroll
    for (int j = 0; j < 4; ++j) sC[tid * 4 + j] = 0.0f;

    // ---- Phase 1: scores -> LDS ------------------------------------------
    const bf16* qrow = Qp + (size_t)(b * SS + q0 + l15) * DD + h * DK;
    v16bf a0 = cat8(*(const v8bf*)(qrow + hi * 16),      *(const v8bf*)(qrow + hi * 16 + 8));
    v16bf a1 = cat8(*(const v8bf*)(qrow + 32 + hi * 16), *(const v8bf*)(qrow + 40 + hi * 16));

    int cbase = w * 256;              // this wave's 256 key columns
    for (int ct = 0; ct < 16; ++ct) {
        int col0 = cbase + ct * 16;
        const bf16* krow = Kp + (size_t)(b * SS + col0 + l15) * DD + h * DK + hi * 16;
        v16bf b0 = cat8(*(const v8bf*)krow,        *(const v8bf*)(krow + 8));
        v16bf b1 = cat8(*(const v8bf*)(krow + 32), *(const v8bf*)(krow + 40));
        v8f acc = v8f{};
        acc = wmma_bf16(a0, b0, acc);
        acc = wmma_bf16(a1, b1, acc);
#pragma unroll
        for (int r = 0; r < 8; ++r)
            sS[(r + hi * 8) * SS + col0 + l15] = acc[r] * INV_SCALE;
    }
    __syncthreads();

    // ---- Phase 2: exact softmax; stream probabilities to d_out -----------
    for (int rr = 0; rr < 2; ++rr) {
        int r = w * 2 + rr;           // row 0..15
        float* srow = sS + r * SS;
        float vmax = -3.4e38f;
        for (int c = lane; c < SS; c += 32) vmax = fmaxf(vmax, srow[c]);
#pragma unroll
        for (int m = 16; m >= 1; m >>= 1) vmax = fmaxf(vmax, __shfl_xor(vmax, m, 32));
        float sum = 0.0f;
        for (int c = lane; c < SS; c += 32) {
            float e = __expf(srow[c] - vmax);
            srow[c] = e;
            sum += e;
        }
#pragma unroll
        for (int m = 16; m >= 1; m >>= 1) sum += __shfl_xor(sum, m, 32);
        float inv = 1.0f / sum;
        float* grow = attn + ((size_t)bh * SS + q0 + r) * SS;
        for (int c = lane; c < SS; c += 32) {
            float p = srow[c] * inv;
            srow[c] = p;
            grow[c] = p;
        }
    }
    __syncthreads();

    // ---- Phase 3: P·V partial over this wave's 256 key columns -----------
    v8f acc4[4] = {v8f{}, v8f{}, v8f{}, v8f{}};
    const bf16* vbase = Vt + (size_t)bh * DK * SS;
    for (int kk = 0; kk < 8; ++kk) {
        int k0 = cbase + kk * 32;
        const float* pr = sS + l15 * SS + k0;
        float4 f0 = *(const float4*)(pr + hi * 16);
        float4 f1 = *(const float4*)(pr + hi * 16 + 4);
        float4 f2 = *(const float4*)(pr + hi * 16 + 8);
        float4 f3 = *(const float4*)(pr + hi * 16 + 12);
        v16bf a = f16x4_to_a(f0, f1, f2, f3);
        const bf16* v0 = vbase + (size_t)(0 * 16 + l15) * SS + k0 + hi * 16;
        const bf16* v1 = vbase + (size_t)(1 * 16 + l15) * SS + k0 + hi * 16;
        const bf16* v2 = vbase + (size_t)(2 * 16 + l15) * SS + k0 + hi * 16;
        const bf16* v3 = vbase + (size_t)(3 * 16 + l15) * SS + k0 + hi * 16;
        v16bf b0 = cat8(*(const v8bf*)v0, *(const v8bf*)(v0 + 8));
        v16bf b1 = cat8(*(const v8bf*)v1, *(const v8bf*)(v1 + 8));
        v16bf b2 = cat8(*(const v8bf*)v2, *(const v8bf*)(v2 + 8));
        v16bf b3 = cat8(*(const v8bf*)v3, *(const v8bf*)(v3 + 8));
        acc4[0] = wmma_bf16(a, b0, acc4[0]);
        acc4[1] = wmma_bf16(a, b1, acc4[1]);
        acc4[2] = wmma_bf16(a, b2, acc4[2]);
        acc4[3] = wmma_bf16(a, b3, acc4[3]);
    }
#pragma unroll
    for (int t = 0; t < 4; ++t)
#pragma unroll
        for (int r = 0; r < 8; ++r)
            atomicAdd(&sC[(r + hi * 8) * 64 + t * 16 + l15], acc4[t][r]);
    __syncthreads();

    // ---- Phase 4: context tile -> bf16 ctx -------------------------------
    {
        int e = tid * 4;              // 1024 floats / 256 threads
        int row = e >> 6, col = e & 63;
        v4bf o;
        o[0] = (bf16)sC[e + 0];
        o[1] = (bf16)sC[e + 1];
        o[2] = (bf16)sC[e + 2];
        o[3] = (bf16)sC[e + 3];
        *(v4bf*)(ctx + (size_t)(b * SS + q0 + row) * DD + h * DK + col) = o;
    }
}

// ---------------------------------------------------------------- launch
extern "C" void kernel_launch(void* const* d_in, const int* in_sizes, int n_in,
                              void* d_out, int out_size, void* d_ws, size_t ws_size,
                              hipStream_t stream) {
    const float* query = (const float*)d_in[0];
    const float* key   = (const float*)d_in[1];
    const float* value = (const float*)d_in[2];
    const float* w_q   = (const float*)d_in[3];
    const float* w_k   = (const float*)d_in[4];
    const float* w_v   = (const float*)d_in[5];
    const float* w_o   = (const float*)d_in[6];

    float* out  = (float*)d_out;                          // [B,S,D]
    float* attn = out + (size_t)BB * SS * DD;             // [B,H,S,S]

    const size_t N_X = (size_t)BB * SS * DD;   // 4,194,304 activations
    const size_t N_W = (size_t)DD * DD;        // 1,048,576 per weight

    char* p = (char*)d_ws;
    auto take = [&](size_t elems) { bf16* r = (bf16*)p; p += elems * sizeof(bf16); return r; };
    bf16* Xq = take(N_X);
    bf16* Xk = take(N_X);
    bf16* Xv = take(N_X);
    bf16* Wq = take(N_W);
    bf16* Wk = take(N_W);
    bf16* Wv = take(N_W);
    bf16* Wo = take(N_W);
    bf16* Qp = take(N_X);
    bf16* Kp = take(N_X);
    bf16* Vp = take(N_X);
    bf16* Vt = take(N_X);   // [B,H,DK,S]
    bf16* Ctx = take(N_X);

    // 1) fp32 -> bf16 conversions
    auto cvt = [&](const float* src, bf16* dst, size_t n) {
        int n4 = (int)(n / 4);
        cvt_f32_bf16_kernel<<<(n4 + 255) / 256, 256, 0, stream>>>(src, dst, n4);
    };
    cvt(query, Xq, N_X);
    cvt(key,   Xk, N_X);
    cvt(value, Xv, N_X);
    cvt(w_q, Wq, N_W);
    cvt(w_k, Wk, N_W);
    cvt(w_v, Wv, N_W);
    cvt(w_o, Wo, N_W);

    // 2) projections  Y = X * W^T   (M=4096, N=1024, K=1024)
    const int M = BB * SS, N = DD, K = DD;
    int gemm_blocks = (M / 16) * (N / 64) / 8;      // 512 blocks (8 waves each)
    gemm_abT_kernel<bf16><<<gemm_blocks, 256, 0, stream>>>(Xq, Wq, Qp, M, N, K);
    gemm_abT_kernel<bf16><<<gemm_blocks, 256, 0, stream>>>(Xk, Wk, Kp, M, N, K);
    gemm_abT_kernel<bf16><<<gemm_blocks, 256, 0, stream>>>(Xv, Wv, Vp, M, N, K);

    // 3) V transpose per head
    {
        size_t n = (size_t)BB * HH * DK * SS;           // 4,194,304
        transpose_v_kernel<<<(int)(n / 256), 256, 0, stream>>>(Vp, Vt);
    }

    // 4) fused scores + softmax + P·V  (132 KB dynamic LDS per workgroup)
    {
        int blocks = BB * HH * (SS / 16);               // 4096
        size_t shbytes = (size_t)(16 * SS + 16 * 64) * sizeof(float);
        fused_attn_kernel<<<blocks, 256, shbytes, stream>>>(Qp, Kp, Vt, attn, Ctx);
    }

    // 5) out = ctx * Wo^T  (fp32 out)
    gemm_abT_kernel<float><<<gemm_blocks, 256, 0, stream>>>(Ctx, Wo, out, M, N, K);
}
